// SharedExpertMLP_9423158247592
// MI455X (gfx1250) — matrix operand order repaired and checked by reading, not compiled, not run on hardware
//
#include <hip/hip_runtime.h>
#include <hip/hip_bf16.h>
#include <math.h>

// ---------------------------------------------------------------------------
// Fused band-LoRA MLP for MI455X (gfx1250, wave32, WMMA bf16 16x16x32).
//   h   = gelu([x | mask(x@A1hat)] @ [w1 ; s*B1stk] + b1)
//   out = h @ w2 + mask(h@A2hat) @ (s*B2stk) + b2
// Weights pre-converted to bf16 column-major in d_ws (per-chunk contiguous),
// staged per H-chunk into LDS with GLOBAL_LOAD_ASYNC_TO_LDS_B128 (ASYNCcnt).
// LDS kept under 160KB so 2 workgroups co-reside per WGP (4 waves/SIMD32),
// letting the other WG's WMMAs cover the exact-erf GELU VALU stretches.
// ---------------------------------------------------------------------------

typedef __bf16 bf16;
typedef __attribute__((ext_vector_type(8)))  bf16  bf16x8;
typedef __attribute__((ext_vector_type(16))) bf16  bf16x16;
typedef __attribute__((ext_vector_type(8)))  float floatx8;
typedef __attribute__((ext_vector_type(4)))  int   intx4;

#define NTOK 262144
#define DIM  192
#define HID  768
#define ODIM 192
#define RR   8
#define LSCALE 2.0f   // alpha/rank = 16/8

#if __has_builtin(__builtin_amdgcn_global_load_async_to_lds_b128) && \
    __has_builtin(__builtin_amdgcn_s_wait_asynccnt)
#define USE_ASYNC 1
#else
#define USE_ASYNC 0
#endif

#define AS1 __attribute__((address_space(1)))
#define AS3 __attribute__((address_space(3)))

// copy 16 bytes global -> LDS, per-lane addresses (async on CDNA5)
__device__ __forceinline__ void stage16(const bf16* g, bf16* l) {
#if USE_ASYNC
  __builtin_amdgcn_global_load_async_to_lds_b128((AS1 intx4*)g, (AS3 intx4*)l, 0, 0);
#else
  *(bf16x8*)l = *(const bf16x8*)g;
#endif
}

__device__ __forceinline__ void wait_async() {
#if USE_ASYNC
  __builtin_amdgcn_s_wait_asynccnt(0);
#endif
}

__device__ __forceinline__ bf16x16 frag_cat(bf16x8 lo, bf16x8 hi) {
  return __builtin_shufflevector(lo, hi, 0,1,2,3,4,5,6,7,8,9,10,11,12,13,14,15);
}

// A fragment (16x32 bf16): lane row M=lane&15; K = kk*32 + (lane>>4)*16 + e,
// contiguous 16 values (matches B). kbase = kk*32 + (lane>>4)*8.
__device__ __forceinline__ bf16x16 load_a_frag(const bf16* __restrict__ row, int kbase) {
  bf16x8 lo = *(const bf16x8*)(row + kbase + (kbase & 8));
  bf16x8 hi = *(const bf16x8*)(row + kbase + (kbase & 8) + 8);
  return frag_cat(lo, hi);
}

// B fragment (32x16 bf16): lane column N=lane&15; K = kk*32 + (lane>>4)*16 + e,
// contiguous 16 values. kbase = kk*32 + (lane>>4)*16.
__device__ __forceinline__ bf16x16 load_b_frag(const bf16* __restrict__ col, int kbase) {
  bf16x8 lo = *(const bf16x8*)(col + kbase);
  bf16x8 hi = *(const bf16x8*)(col + kbase + 8);
  return frag_cat(lo, hi);
}

__device__ __forceinline__ floatx8 wmma_bf16(bf16x16 a, bf16x16 b, floatx8 c) {
  return __builtin_amdgcn_wmma_f32_16x16x32_bf16(false, a, false, b, (short)0, c, false, false);
}

__device__ __forceinline__ float gelu_exact(float v) {
  return 0.5f * v * (1.0f + erff(v * 0.70710678118654752f));
}

// ----------------------------- prep kernels --------------------------------
// W1catT[n*256 + k]: k<192 -> w1[k][n]; k>=192 -> s*B1[(k-192)][n]  (b*8+r = k-192)
// H-chunk ch occupies the contiguous block [ch*64*256, (ch+1)*64*256).
__global__ void prep_w1cat(const float* __restrict__ w1, const float* __restrict__ B1,
                           bf16* __restrict__ dst) {
  int idx = blockIdx.x * 256 + threadIdx.x;
  if (idx >= HID * 256) return;
  int n = idx >> 8, k = idx & 255;
  float v = (k < DIM) ? w1[k * HID + n] : LSCALE * B1[(k - DIM) * HID + n];
  dst[idx] = (bf16)v;
}

// W2C[((ch*192)+n)*64 + kk] = w2[ch*64+kk][n]   (per-chunk contiguous 24KB blocks)
__global__ void prep_w2c(const float* __restrict__ w2, bf16* __restrict__ dst) {
  int idx = blockIdx.x * 256 + threadIdx.x;
  if (idx >= 12 * ODIM * 64) return;
  int kk = idx & 63, t = idx >> 6;
  int n = t % ODIM, ch = t / ODIM;
  dst[idx] = (bf16)w2[(ch * 64 + kk) * ODIM + n];
}

// A2C[((ch*64)+n)*64 + kk] = A2[b][ch*64+kk][r], b=n>>3, r=n&7  (8KB blocks)
__global__ void prep_a2c(const float* __restrict__ A2, bf16* __restrict__ dst) {
  int idx = blockIdx.x * 256 + threadIdx.x;
  if (idx >= 12 * 64 * 64) return;
  int kk = idx & 63, t = idx >> 6;
  int n = t & 63, ch = t >> 6;
  int b = n >> 3, r = n & 7;
  dst[idx] = (bf16)A2[(b * HID + ch * 64 + kk) * RR + r];
}

// A1hatT[n*192 + k] = A1[b][k][r], b = n>>3, r = n&7
__global__ void prep_a1hat(const float* __restrict__ A1, bf16* __restrict__ dst) {
  int idx = blockIdx.x * 256 + threadIdx.x;
  if (idx >= 64 * DIM) return;
  int n = idx / DIM, k = idx - n * DIM;
  int b = n >> 3, r = n & 7;
  dst[idx] = (bf16)A1[(b * DIM + k) * RR + r];
}

// B2stackT[n*64 + k] = s*B2[k][n]   (b*8+r = k)
__global__ void prep_b2stack(const float* __restrict__ B2, bf16* __restrict__ dst) {
  int idx = blockIdx.x * 256 + threadIdx.x;
  if (idx >= ODIM * 64) return;
  int n = idx >> 6, k = idx & 63;
  dst[idx] = (bf16)(LSCALE * B2[k * ODIM + n]);
}

// --------------------- async weight staging (per H-chunk) ------------------
__device__ __forceinline__ void stage_w1(int ch, int wave, int lane,
                                         const bf16* __restrict__ W1catT, bf16* w1b) {
#pragma unroll
  for (int j = 0; j < 8; ++j) {            // one 512B column per wave-op
    int col = wave * 8 + j;
    stage16(W1catT + (ch * 64 + col) * 256 + lane * 8, w1b + col * 264 + lane * 8);
  }
}

__device__ __forceinline__ void stage_w2a2(int ch, int wave, int lane,
                                           const bf16* __restrict__ W2C,
                                           const bf16* __restrict__ A2C,
                                           bf16* w2b, bf16* a2b) {
  const int sub = lane >> 3, part = lane & 7;
#pragma unroll
  for (int j = 0; j < 6; ++j) {            // W2: 4 columns (128B each) per wave-op
    int col = (wave * 6 + j) * 4 + sub;
    stage16(W2C + (ch * ODIM + col) * 64 + part * 8, w2b + col * 72 + part * 8);
  }
#pragma unroll
  for (int j = 0; j < 2; ++j) {            // A2hat: 4 columns per wave-op
    int col = (wave * 2 + j) * 4 + sub;
    stage16(A2C + (ch * 64 + col) * 64 + part * 8, a2b + col * 72 + part * 8);
  }
}

// ------------------------------ main kernel --------------------------------
__global__ __launch_bounds__(256) void fused_mlp_kernel(
    const float* __restrict__ x, const int* __restrict__ band_idx,
    const float* __restrict__ b1, const float* __restrict__ b2,
    const bf16* __restrict__ W1catT, const bf16* __restrict__ W2C,
    const bf16* __restrict__ A1hatT, const bf16* __restrict__ A2C,
    const bf16* __restrict__ B2stackT, float* __restrict__ out) {
  __shared__ bf16 XV[64][264];        // [tok][0..191 x | 192..255 masked u1], padded
  __shared__ bf16 HB[64][72];         // gelu'd h chunk / v2 in phase 3
  __shared__ bf16 W1buf[64][264];     // W1cat chunk (single, refilled mid-iteration)
  __shared__ bf16 W2buf[2][192][72];  // double-buffered w2 chunk
  __shared__ bf16 A2buf[2][64][72];   // double-buffered A2hat chunk
  __shared__ int  bands[64];
  // total ~151 KB -> two workgroups per WGP

  const int tid   = threadIdx.x;
  const int lane  = tid & 31;
  const int wave  = tid >> 5;
  const int rblk  = wave & 3;    // 16-token row block
  const int chalf = wave >> 2;   // column half
  const int lr    = lane & 15;
  const int lh    = lane >> 4;
  const int tokBase = blockIdx.x * 64;

  // kick off async staging of chunk 0 ASAP (overlaps x conversion + phase 1)
  stage_w1(0, wave, lane, W1catT, &W1buf[0][0]);
  stage_w2a2(0, wave, lane, W2C, A2C, &W2buf[0][0][0], &A2buf[0][0][0]);

  if (tid < 64) bands[tid] = band_idx[tokBase + tid];

  { // stage x tile -> LDS as bf16 (coalesced float4 reads)
    const int m = tid >> 2, q = tid & 3;
    const float4* src = (const float4*)(x + (size_t)(tokBase + m) * DIM) + q * 12;
    bf16* dst = &XV[m][q * 48];
#pragma unroll
    for (int j = 0; j < 12; ++j) {
      float4 v = src[j];
      dst[j * 4 + 0] = (bf16)v.x; dst[j * 4 + 1] = (bf16)v.y;
      dst[j * 4 + 2] = (bf16)v.z; dst[j * 4 + 3] = (bf16)v.w;
    }
  }
  __syncthreads();

  const bf16* arow = &XV[rblk * 16 + lr][0];

  // ---- phase 1: u1_all = x @ A1hat (K=192), mask by band, append to XV ----
  {
    floatx8 u1[2] = {{0,0,0,0,0,0,0,0}, {0,0,0,0,0,0,0,0}};
#pragma unroll
    for (int kk = 0; kk < 6; ++kk) {
      bf16x16 a = load_a_frag(arow, kk * 32 + lh * 8);
#pragma unroll
      for (int t = 0; t < 2; ++t) {
        int n_g = chalf * 32 + t * 16 + lr;
        bf16x16 b = load_b_frag(A1hatT + n_g * DIM, kk * 32 + lh * 16);
        u1[t] = wmma_bf16(a, b, u1[t]);
      }
    }
#pragma unroll
    for (int t = 0; t < 2; ++t) {
      int n_g = chalf * 32 + t * 16 + lr;
      int bn  = n_g >> 3;
#pragma unroll
      for (int i = 0; i < 8; ++i) {
        int m = rblk * 16 + lh * 8 + i;
        float v = (bands[m] == bn) ? u1[t][i] : 0.0f;
        XV[m][DIM + n_g] = (bf16)v;
      }
    }
  }

  // ---- phase 2: stream H in 12 chunks of 64 ----
  floatx8 outacc[6];
#pragma unroll
  for (int j = 0; j < 6; ++j) outacc[j] = (floatx8){0,0,0,0,0,0,0,0};
  floatx8 u2[2] = {{0,0,0,0,0,0,0,0}, {0,0,0,0,0,0,0,0}};

  for (int ch = 0; ch < 12; ++ch) {
    const int cur = ch & 1;
    const bf16* w2b = &W2buf[cur][0][0];
    const bf16* a2b = &A2buf[cur][0][0];

    wait_async();      // chunk ch (W1 mid-prev-iter, W2/A2 top-prev-iter) resident
    __syncthreads();   // + all waves done with buffers being overwritten below
                       //   (for ch==0 this also publishes XV/phase-1)

    if (ch + 1 < 12)   // W2/A2 for ch+1: free buffer, overlaps whole iteration
      stage_w2a2(ch + 1, wave, lane, W2C, A2C,
                 &W2buf[1 - cur][0][0], &A2buf[1 - cur][0][0]);

    // GEMM1: h_chunk = [x|v1] @ W1cat[:, chunk]   (K = 256)
    floatx8 hacc[2] = {{0,0,0,0,0,0,0,0}, {0,0,0,0,0,0,0,0}};
#pragma unroll
    for (int kk = 0; kk < 8; ++kk) {
      bf16x16 a = load_a_frag(arow, kk * 32 + lh * 8);
#pragma unroll
      for (int t = 0; t < 2; ++t) {
        int cl = chalf * 32 + t * 16 + lr;
        bf16x16 b = load_b_frag(&W1buf[cl][0], kk * 32 + lh * 16);
        hacc[t] = wmma_bf16(a, b, hacc[t]);
      }
    }
    // bias + exact GELU, write bf16 h chunk to LDS
#pragma unroll
    for (int t = 0; t < 2; ++t) {
      int ncol = chalf * 32 + t * 16 + lr;
      float bias = b1[ch * 64 + ncol];
#pragma unroll
      for (int i = 0; i < 8; ++i) {
        float v = gelu_exact(hacc[t][i] + bias);
        HB[rblk * 16 + lh * 8 + i][ncol] = (bf16)v;
      }
    }
    __syncthreads();   // HB ready; also: all GEMM1 reads of W1buf retired

    if (ch + 1 < 12)   // refill W1buf for ch+1; overlaps GEMM2a/2b below
      stage_w1(ch + 1, wave, lane, W1catT, &W1buf[0][0]);

    // GEMM2a: out += h_chunk @ w2[chunk,:]; GEMM2b: u2 += h_chunk @ A2hat[chunk,:]
    const bf16* hrow = &HB[rblk * 16 + lr][0];
#pragma unroll
    for (int kk = 0; kk < 2; ++kk) {
      bf16x16 a = load_a_frag(hrow, kk * 32 + lh * 8);
      int kb = kk * 32 + lh * 16;
#pragma unroll
      for (int j = 0; j < 6; ++j) {
        int n_g = chalf * 96 + j * 16 + lr;
        bf16x16 b = load_b_frag(w2b + n_g * 72, kb);
        outacc[j] = wmma_bf16(a, b, outacc[j]);
      }
#pragma unroll
      for (int t = 0; t < 2; ++t) {
        int cl = chalf * 32 + t * 16 + lr;
        bf16x16 b = load_b_frag(a2b + cl * 72, kb);
        u2[t] = wmma_bf16(a, b, u2[t]);
      }
    }
  }
  __syncthreads();   // all waves done reading HB before it becomes v2

  // ---- phase 3: mask u2 -> v2, out += v2 @ (s*B2stack), add b2, store ----
#pragma unroll
  for (int t = 0; t < 2; ++t) {
    int n_g = chalf * 32 + t * 16 + lr;
    int bn  = n_g >> 3;
#pragma unroll
    for (int i = 0; i < 8; ++i) {
      int m = rblk * 16 + lh * 8 + i;
      float v = (bands[m] == bn) ? u2[t][i] : 0.0f;
      HB[m][n_g] = (bf16)v;
    }
  }
  __syncthreads();
  {
    const bf16* vrow = &HB[rblk * 16 + lr][0];
#pragma unroll
    for (int kk = 0; kk < 2; ++kk) {
      bf16x16 a = load_a_frag(vrow, kk * 32 + lh * 8);
#pragma unroll
      for (int j = 0; j < 6; ++j) {
        int n_g = chalf * 96 + j * 16 + lr;
        bf16x16 b = load_b_frag(B2stackT + n_g * 64, kk * 32 + lh * 16);
        outacc[j] = wmma_bf16(a, b, outacc[j]);
      }
    }
  }
#pragma unroll
  for (int j = 0; j < 6; ++j) {
    int n_g = chalf * 96 + j * 16 + lr;
    float bias = b2[n_g];
#pragma unroll
    for (int i = 0; i < 8; ++i) {
      int m = rblk * 16 + lh * 8 + i;
      out[(size_t)(tokBase + m) * ODIM + n_g] = outacc[j][i] + bias;
    }
  }
}

// ------------------------------- launcher ----------------------------------
extern "C" void kernel_launch(void* const* d_in, const int* in_sizes, int n_in,
                              void* d_out, int out_size, void* d_ws, size_t ws_size,
                              hipStream_t stream) {
  const float* x    = (const float*)d_in[0];
  const int*   band = (const int*)  d_in[1];
  const float* w1   = (const float*)d_in[2];
  const float* b1   = (const float*)d_in[3];
  const float* w2   = (const float*)d_in[4];
  const float* b2   = (const float*)d_in[5];
  const float* A1   = (const float*)d_in[6];
  const float* B1   = (const float*)d_in[7];
  const float* A2   = (const float*)d_in[8];
  const float* B2   = (const float*)d_in[9];
  float* out = (float*)d_out;

  bf16* ws       = (bf16*)d_ws;
  bf16* W1catT   = ws;               // 768*256    = 196608
  bf16* W2C      = ws + 196608;      // 12*192*64  = 147456
  bf16* A2C      = ws + 344064;      // 12*64*64   =  49152
  bf16* A1hatT   = ws + 393216;      // 64*192     =  12288
  bf16* B2stackT = ws + 405504;      // 192*64     =  12288   (total 835584 B)

  prep_w1cat  <<<(HID * 256 + 255) / 256, 256, 0, stream>>>(w1, B1, W1catT);
  prep_w2c    <<<(12 * ODIM * 64 + 255) / 256, 256, 0, stream>>>(w2, W2C);
  prep_a2c    <<<(12 * 64 * 64 + 255) / 256, 256, 0, stream>>>(A2, A2C);
  prep_a1hat  <<<(64 * DIM + 255) / 256, 256, 0, stream>>>(A1, A1hatT);
  prep_b2stack<<<(ODIM * 64 + 255) / 256, 256, 0, stream>>>(B2, B2stackT);

  fused_mlp_kernel<<<NTOK / 64, 256, 0, stream>>>(
      x, band, b1, b2, W1catT, W2C, A1hatT, A2C, B2stackT, out);
}
